// TreeGCN_69810398429649
// MI455X (gfx1250) — compile-verified, matrix operand order checked by reading, not run on hardware
//
#include <hip/hip_runtime.h>
#include <hip/hip_bf16.h>
#include <math.h>

typedef __attribute__((ext_vector_type(16))) _Float16 v16h;
typedef __attribute__((ext_vector_type(8)))  float    v8f;

// ---------------------------------------------------------------------------
// CDNA5 WMMA fragment helpers (wave32, v_wmma_f32_16x16x32_f16)
// A 16x32 f16 (ISA 7.12.2): lane<16: M=lane, K in {0..7,16..23};
// lane>=16: same M, K in {8..15,24..31}. half idx: vgpr=idx>>1, bit=idx&1,
// grp=vgpr>>2, sub=vgpr&3 -> k = grp*16 + (lane>=16?8:0) + sub*2 + bit.
// ---------------------------------------------------------------------------
__device__ __forceinline__ v16h load_a_frag_lds(const float* At, int ld, int kbase) {
    const int lane = threadIdx.x & 31;
    const int m    = lane & 15;
    const int hi8  = (lane >> 4) << 3;
    const float* row = At + m * ld + kbase;
    v16h a;
#pragma unroll
    for (int j = 0; j < 8; ++j) {             // VGPR j holds k, k+1 (consecutive)
        const int grp = j >> 2, sub = j & 3;
        const int k   = grp * 16 + hi8 + sub * 2;
        const float2 f2 = *(const float2*)(row + k);   // ds_load_b64
        a[2 * j]     = (_Float16)f2.x;
        a[2 * j + 1] = (_Float16)f2.y;
    }
    return a;
}

// B fragments come from a pre-packed f16 buffer laid out
// [ntile][kchunk][lane(32)][16 halves]  -> one aligned 32B load per lane.
__device__ __forceinline__ v16h load_b_frag_packed(const _Float16* P, int nt, int kc,
                                                   int kch) {
    const int lane = threadIdx.x & 31;
    const size_t off = (((size_t)(nt * kch + kc) * 32) + lane) << 4;
    return *(const v16h*)(P + off);            // 2x global_load_b128
}

// C/D 16x16 f32: VGPR v holds M=v (lanes 0-15) / M=v+8 (lanes 16-31), N=lane&15.
__device__ __forceinline__ void store_d_lds(float* Ct, int ldc, int nbase, v8f d) {
    const int lane = threadIdx.x & 31;
    const int n    = nbase + (lane & 15);
    const int mh   = (lane >> 4) * 8;
#pragma unroll
    for (int v = 0; v < 8; ++v) Ct[(mh + v) * ldc + n] = d[v];
}

// ---------------------------------------------------------------------------
// Weight pre-pack: B element (k,n) at B[k*sBk + n*sBn] -> fragment order f16.
// ---------------------------------------------------------------------------
__global__ void pack_weight_kernel(const float* __restrict__ B, int sBk, int sBn,
                                   int Kact, int Nact, int kch, int ntiles,
                                   _Float16* __restrict__ P) {
    const int total = ntiles * kch * 32 * 16;
    int i = blockIdx.x * blockDim.x + threadIdx.x;
    if (i >= total) return;
    const int idx  = i & 15;
    const int lane = (i >> 4) & 31;
    const int kc   = (i >> 9) % kch;
    const int nt   = (i >> 9) / kch;
    const int vg = idx >> 1, bit = idx & 1;
    const int grp = vg >> 2, sub = vg & 3;
    const int k = kc * 32 + grp * 16 + ((lane >> 4) << 3) + sub * 2 + bit;
    const int n = nt * 16 + (lane & 15);
    float v = 0.0f;
    if (k < Kact && n < Nact) v = B[k * sBk + n * sBn];
    P[i] = (_Float16)v;
}

// ---------------------------------------------------------------------------
// Fused GRU: one block = 16 nodes, h resident in LDS across all L steps.
// Per step: wx = x @ W_ih^T, gh = h @ W_hh^T (WMMA, K pad 128, N pad 304),
// then gate math. No Wx intermediate in HBM.
// ---------------------------------------------------------------------------
#define GRU_D  100
#define GRU_H  100
#define GRU_G  300
#define GRU_KP 128
#define GRU_KC 4
#define GRU_NP 304
#define GRU_NT 19

__global__ __launch_bounds__(256)
void gru_fused_kernel(const int* __restrict__ feat, const float* __restrict__ emb,
                      const float* __restrict__ h0,
                      const _Float16* __restrict__ pIH, const _Float16* __restrict__ pHH,
                      const float* __restrict__ b_ih, const float* __restrict__ b_hh,
                      float* __restrict__ hn, int N, int L) {
    __shared__ float xT[16 * GRU_KP];
    __shared__ float hT[16 * GRU_KP];
    __shared__ float wxT[16 * GRU_NP];
    __shared__ float ghT[16 * GRU_NP];

    const int tid      = threadIdx.x;
    const int wave     = tid >> 5;
    const int nwaves   = blockDim.x >> 5;
    const int nodeBase = blockIdx.x * 16;

    // init hidden tile; zero K-pad columns of both tiles once
    for (int i = tid; i < 16 * GRU_KP; i += blockDim.x) {
        const int m = i / GRU_KP, c = i % GRU_KP;
        const int node = nodeBase + m;
        hT[i] = (node < N && c < GRU_H) ? h0[node * GRU_H + c] : 0.0f;
        xT[i] = 0.0f;
    }
    __syncthreads();

    for (int l = 0; l < L; ++l) {
        // embedding gather (only real columns; pad stays zero)
        for (int i = tid; i < 16 * GRU_D; i += blockDim.x) {
            const int m = i / GRU_D, c = i % GRU_D;
            const int node = nodeBase + m;
            float v = 0.0f;
            if (node < N) {
                const int tok = feat[node * L + l];
                v = emb[tok * GRU_D + c];
                if (c == 0 && l + 1 < L) {
                    const int tok2 = feat[node * L + l + 1];
                    __builtin_prefetch(&emb[tok2 * GRU_D], 0, 1);
                }
            }
            xT[m * GRU_KP + c] = v;
        }
        __syncthreads();

        // dual WMMA GEMMs: wx = x*W_ih^T ; gh = h*W_hh^T
        for (int nt = wave; nt < GRU_NT; nt += nwaves) {
            const int nbase = nt * 16;
            v8f accW = {};
            v8f accG = {};
#pragma unroll
            for (int kc = 0; kc < GRU_KC; ++kc) {
                const int kb = kc * 32;
                v16h aX = load_a_frag_lds(xT, GRU_KP, kb);
                v16h aH = load_a_frag_lds(hT, GRU_KP, kb);
                v16h bI = load_b_frag_packed(pIH, nt, kc, GRU_KC);
                v16h bH = load_b_frag_packed(pHH, nt, kc, GRU_KC);
                accW = __builtin_amdgcn_wmma_f32_16x16x32_f16(false, aX, false, bI,
                                                              (short)0, accW, false, false);
                accG = __builtin_amdgcn_wmma_f32_16x16x32_f16(false, aH, false, bH,
                                                              (short)0, accG, false, false);
            }
            store_d_lds(wxT, GRU_NP, nbase, accW);
            store_d_lds(ghT, GRU_NP, nbase, accG);
        }
        __syncthreads();

        // gate math + hidden update (torch GRU cell)
        for (int i = tid; i < 16 * GRU_H; i += blockDim.x) {
            const int m = i / GRU_H, j = i % GRU_H;
            const float ir  = wxT[m * GRU_NP + j]             + b_ih[j];
            const float iz  = wxT[m * GRU_NP + GRU_H + j]     + b_ih[GRU_H + j];
            const float inn = wxT[m * GRU_NP + 2 * GRU_H + j] + b_ih[2 * GRU_H + j];
            const float hr  = ghT[m * GRU_NP + j]             + b_hh[j];
            const float hz  = ghT[m * GRU_NP + GRU_H + j]     + b_hh[GRU_H + j];
            const float hnn = ghT[m * GRU_NP + 2 * GRU_H + j] + b_hh[2 * GRU_H + j];
            const float r = 1.0f / (1.0f + __expf(-(ir + hr)));
            const float z = 1.0f / (1.0f + __expf(-(iz + hz)));
            const float n = tanhf(inn + r * hnn);
            const float hp = hT[m * GRU_KP + j];
            hT[m * GRU_KP + j] = (1.0f - z) * n + z * hp;
        }
        __syncthreads();
    }

    for (int i = tid; i < 16 * GRU_H; i += blockDim.x) {
        const int m = i / GRU_H, j = i % GRU_H;
        const int node = nodeBase + m;
        if (node < N) hn[node * GRU_H + j] = hT[m * GRU_KP + j];
    }
}

// ---------------------------------------------------------------------------
// Generic WMMA GEMM: C[M,Nact] = A[M,Kact] (f32 row-major) * packed-B (f16)
// ---------------------------------------------------------------------------
#define GEMM_KPMAX 224

__global__ __launch_bounds__(256)
void gemm_wmma_kernel(const float* __restrict__ A, int lda, int Kact,
                      const _Float16* __restrict__ Pb, int Nact,
                      float* __restrict__ C, int ldc, int M) {
    __shared__ float aT[16 * GEMM_KPMAX];
    const int tid    = threadIdx.x;
    const int wave   = tid >> 5;
    const int nwaves = blockDim.x >> 5;
    const int mbase  = blockIdx.x * 16;
    const int kch    = (Kact + 31) >> 5;
    const int ld     = kch * 32;

    for (int i = tid; i < 16 * ld; i += blockDim.x) {
        const int m = i / ld, c = i % ld;
        const int row = mbase + m;
        aT[i] = (row < M && c < Kact) ? A[row * lda + c] : 0.0f;
    }
    __syncthreads();

    const int ntiles = (Nact + 15) >> 4;
    for (int nt = wave; nt < ntiles; nt += nwaves) {
        const int nbase = nt * 16;
        v8f acc = {};
        for (int kc = 0; kc < kch; ++kc) {
            v16h a = load_a_frag_lds(aT, ld, kc * 32);
            v16h b = load_b_frag_packed(Pb, nt, kc, kch);
            acc = __builtin_amdgcn_wmma_f32_16x16x32_f16(false, a, false, b,
                                                         (short)0, acc, false, false);
        }
        const int lane = tid & 31;
        const int n    = nbase + (lane & 15);
        const int mh   = (lane >> 4) * 8;
#pragma unroll
        for (int v = 0; v < 8; ++v) {
            const int row = mbase + mh + v;
            if (row < M && n < Nact) C[row * ldc + n] = acc[v];
        }
    }
}

// ---------------------------------------------------------------------------
// GCN support kernels
// ---------------------------------------------------------------------------
__global__ void zero_f32_kernel(float* p, int n) {
    int i = blockIdx.x * blockDim.x + threadIdx.x;
    if (i < n) p[i] = 0.0f;
}

__global__ void degree_kernel(const int* __restrict__ dst, float* __restrict__ cnt, int E) {
    int e = blockIdx.x * blockDim.x + threadIdx.x;
    if (e < E) atomicAdd(&cnt[dst[e]], 1.0f);
}

// out[n,f] = xw[n,f] * (1/deg[n]) + bias[f]   (self-loop term, deg incl. self)
__global__ void gcn_self_kernel(const float* __restrict__ xw, const float* __restrict__ cnt,
                                const float* __restrict__ bias, float* __restrict__ out,
                                int N, int F) {
    int i = blockIdx.x * blockDim.x + threadIdx.x;
    if (i >= N * F) return;
    const int n = i / F, f = i % F;
    const float dinv2 = 1.0f / (cnt[n] + 1.0f);
    out[i] = xw[i] * dinv2 + bias[f];
}

// out[dst,f] += xw[src,f] * rsqrt(deg[src]) * rsqrt(deg[dst])
__global__ void gcn_scatter_kernel(const float* __restrict__ xw, const float* __restrict__ cnt,
                                   const int* __restrict__ src, const int* __restrict__ dst,
                                   float* __restrict__ out, int E, int F) {
    int i = blockIdx.x * blockDim.x + threadIdx.x;
    if (i >= E * F) return;
    const int e = i / F, f = i % F;
    const int s = src[e], t = dst[e];
    const float coef = rsqrtf(cnt[s] + 1.0f) * rsqrtf(cnt[t] + 1.0f);
    atomicAdd(&out[t * F + f], xw[s * F + f] * coef);
}

// xcat[n, 0:100] = relu(xc1[n]);  xcat[n, 100:200] = relu(hn[indices[n]])
__global__ void concat_relu_kernel(const float* __restrict__ xc1, const float* __restrict__ hn,
                                   const int* __restrict__ idx, float* __restrict__ xcat,
                                   int N, int H) {
    int i = blockIdx.x * blockDim.x + threadIdx.x;
    if (i >= N * 2 * H) return;
    const int n = i / (2 * H), j = i % (2 * H);
    float v;
    if (j < H) v = xc1[n * H + j];
    else       v = hn[idx[n] * H + (j - H)];
    xcat[i] = fmaxf(v, 0.0f);
}

// out[n, 0:100] = relu(xc2[n]);  out[n, 100:200] = xc1[indices[n]]
__global__ void final_kernel(const float* __restrict__ xc2, const float* __restrict__ xc1,
                             const int* __restrict__ idx, float* __restrict__ out,
                             int N, int H) {
    int i = blockIdx.x * blockDim.x + threadIdx.x;
    if (i >= N * 2 * H) return;
    const int n = i / (2 * H), j = i % (2 * H);
    if (j < H) out[i] = fmaxf(xc2[n * H + j], 0.0f);
    else       out[i] = xc1[idx[n] * H + (j - H)];
}

// ---------------------------------------------------------------------------
extern "C" void kernel_launch(void* const* d_in, const int* in_sizes, int n_in,
                              void* d_out, int out_size, void* d_ws, size_t ws_size,
                              hipStream_t stream) {
    const int*   feat    = (const int*)d_in[0];
    const int*   edge    = (const int*)d_in[1];
    const int*   indices = (const int*)d_in[2];
    const float* h0      = (const float*)d_in[3];
    const float* emb     = (const float*)d_in[4];
    const float* W_ih    = (const float*)d_in[5];
    const float* W_hh    = (const float*)d_in[6];
    const float* b_ih    = (const float*)d_in[7];
    const float* b_hh    = (const float*)d_in[8];
    const float* W1      = (const float*)d_in[9];
    const float* b1      = (const float*)d_in[10];
    const float* W2      = (const float*)d_in[11];
    const float* b2      = (const float*)d_in[12];

    const int N = in_sizes[2];
    const int E = in_sizes[1] / 2;
    const int L = in_sizes[0] / N;
    const int H = 100;

    float* out  = (float*)d_out;
    float* ws   = (float*)d_ws;
    float* hn   = ws;                        // N*H
    float* xw1  = hn   + (size_t)N * H;      // N*H
    float* xc1  = xw1  + (size_t)N * H;      // N*H
    float* xcat = xc1  + (size_t)N * H;      // N*2H
    float* xw2  = xcat + (size_t)N * 2 * H;  // N*H
    float* xc2  = xw2  + (size_t)N * H;      // N*H
    float* cnt  = xc2  + (size_t)N * H;      // N

    // packed f16 weight fragments (32B-aligned: float region is mult. of 32B)
    _Float16* pIH = (_Float16*)(cnt + N);        // 19*4*512
    _Float16* pHH = pIH + 19 * 4 * 512;          // 19*4*512
    _Float16* pW1 = pHH + 19 * 4 * 512;          // 7*4*512
    _Float16* pW2 = pW1 + 7 * 4 * 512;           // 7*7*512

    const int* srcArr = edge;
    const int* dstArr = edge + E;

    const dim3 blk(256);
    const int mblocks = (N + 15) / 16;

    // 0) pack weights into WMMA fragment order (f16)
    {   // W_ih: B(k=d,n=g) = W_ih[g*100+d] -> sBk=1, sBn=100; K=100 (kch 4), N=300 (nt 19)
        int tot = 19 * 4 * 512;
        pack_weight_kernel<<<(tot + 255) / 256, blk, 0, stream>>>(W_ih, 1, GRU_D,
                                                                  GRU_D, GRU_G, 4, 19, pIH);
        pack_weight_kernel<<<(tot + 255) / 256, blk, 0, stream>>>(W_hh, 1, GRU_H,
                                                                  GRU_H, GRU_G, 4, 19, pHH);
        int t1 = 7 * 4 * 512;  // W1: B(k,n)=W1[k*100+n] -> sBk=100, sBn=1; K=100, N=100
        pack_weight_kernel<<<(t1 + 255) / 256, blk, 0, stream>>>(W1, H, 1, H, H, 4, 7, pW1);
        int t2 = 7 * 7 * 512;  // W2: K=200 (kch 7), N=100
        pack_weight_kernel<<<(t2 + 255) / 256, blk, 0, stream>>>(W2, H, 1, 2 * H, H, 7, 7, pW2);
    }

    // 1) fused GRU (WMMA) -> hn
    gru_fused_kernel<<<mblocks, blk, 0, stream>>>(feat, emb, h0, pIH, pHH,
                                                  b_ih, b_hh, hn, N, L);

    // 2) degree (dst-side, self-loop applied at use)
    zero_f32_kernel<<<(N + 255) / 256, blk, 0, stream>>>(cnt, N);
    degree_kernel<<<(E + 255) / 256, blk, 0, stream>>>(dstArr, cnt, E);

    // 3) GCNConv1: xw1 = hn @ W1 (WMMA); xc1 = self + b1; scatter edges
    gemm_wmma_kernel<<<mblocks, blk, 0, stream>>>(hn, H, H, pW1, H, xw1, H, N);
    gcn_self_kernel<<<(N * H + 255) / 256, blk, 0, stream>>>(xw1, cnt, b1, xc1, N, H);
    gcn_scatter_kernel<<<(E * H + 255) / 256, blk, 0, stream>>>(xw1, cnt, srcArr, dstArr,
                                                                xc1, E, H);

    // 4) relu(concat(xc1, hn[indices]))
    concat_relu_kernel<<<(N * 2 * H + 255) / 256, blk, 0, stream>>>(xc1, hn, indices,
                                                                    xcat, N, H);

    // 5) GCNConv2: xw2 = xcat @ W2 (WMMA, K=200); xc2 = self + b2; scatter
    gemm_wmma_kernel<<<mblocks, blk, 0, stream>>>(xcat, 2 * H, 2 * H, pW2, H, xw2, H, N);
    gcn_self_kernel<<<(N * H + 255) / 256, blk, 0, stream>>>(xw2, cnt, b2, xc2, N, H);
    gcn_scatter_kernel<<<(E * H + 255) / 256, blk, 0, stream>>>(xw2, cnt, srcArr, dstArr,
                                                                xc2, E, H);

    // 6) out = concat(relu(xc2), xc1[indices])
    final_kernel<<<(N * 2 * H + 255) / 256, blk, 0, stream>>>(xc2, xc1, indices, out, N, H);
}
